// CXLoss_72284299591793
// MI455X (gfx1250) — matrix operand order repaired and checked by reading, not compiled, not run on hardware
//
#include <hip/hip_runtime.h>
#include <math.h>

#define HW     4096
#define C_DIM  512
#define N_IMG  4
#define KC     16          // 512 / 32 k-chunks
#define CX_EPS 1e-5f

typedef __attribute__((ext_vector_type(16))) __bf16       bf16x16;
typedef __attribute__((ext_vector_type(8)))  float        f32x8;
typedef __attribute__((ext_vector_type(4)))  unsigned int u32x4;

union Frag {
    bf16x16 v;
    u32x4   q[2];
};

__device__ __forceinline__ unsigned short f2bf(float f) {
    unsigned u = __float_as_uint(f);
    unsigned r = ((u >> 16) & 1u) + 0x7FFFu;   // round-to-nearest-even
    return (unsigned short)((u + r) >> 16);
}

// ---------------------------------------------------------------------------
// 1) mu[img][c] = mean over HW of target
// ---------------------------------------------------------------------------
__global__ void __launch_bounds__(256) mu_kernel(const float* __restrict__ target,
                                                 float* __restrict__ mu) {
    int b = blockIdx.x;                       // img*512 + c
    const float* src = target + (size_t)b * HW;
    int tid = threadIdx.x;
    float s = 0.f;
    for (int i = tid; i < HW; i += 256) s += src[i];
    __shared__ float red[256];
    red[tid] = s; __syncthreads();
    for (int off = 128; off > 0; off >>= 1) {
        if (tid < off) red[tid] += red[tid + off];
        __syncthreads();
    }
    if (tid == 0) mu[b] = red[0] * (1.0f / HW);
}

// ---------------------------------------------------------------------------
// 2) center by mu, L2-normalize along C, pack bf16 as [img][k/32][pix][k%32]
// ---------------------------------------------------------------------------
__global__ void __launch_bounds__(256) normpack_kernel(
        const float* __restrict__ pred, const float* __restrict__ target,
        const float* __restrict__ mu,
        unsigned short* __restrict__ ppack, unsigned short* __restrict__ tpack) {
    int idx = blockIdx.x * 256 + threadIdx.x;   // img*HW + pix
    int img = idx >> 12;
    int pix = idx & (HW - 1);
    const float* P = pred   + (size_t)img * C_DIM * HW + pix;
    const float* T = target + (size_t)img * C_DIM * HW + pix;
    const float* M = mu + img * C_DIM;
    float sp = 0.f, st = 0.f;
    for (int c = 0; c < C_DIM; ++c) {
        float m = M[c];
        float a = P[(size_t)c * HW] - m;
        float b = T[(size_t)c * HW] - m;
        sp += a * a; st += b * b;
    }
    float ip = 1.f / (sqrtf(sp) + CX_EPS);
    float it = 1.f / (sqrtf(st) + CX_EPS);
    unsigned short* pp = ppack + (size_t)img * C_DIM * HW;
    unsigned short* tp = tpack + (size_t)img * C_DIM * HW;
    for (int c = 0; c < C_DIM; ++c) {
        float m = M[c];
        size_t o = ((size_t)(c >> 5) * HW + pix) * 32 + (c & 31);
        pp[o] = f2bf((P[(size_t)c * HW] - m) * ip);
        tp[o] = f2bf((T[(size_t)c * HW] - m) * it);
    }
}

// Fragment loads matching CDNA5 16-bit WMMA layouts (pack stride 32 per pixel)
__device__ __forceinline__ Frag loadA(const unsigned short* base, int g) {
    // lane-half g: K in {8g..8g+7, 16+8g..16+8g+7}
    Frag a;
    a.q[0] = *(const u32x4*)(base + 8 * g);
    a.q[1] = *(const u32x4*)(base + 16 + 8 * g);
    return a;
}
__device__ __forceinline__ Frag loadB(const unsigned short* base, int g) {
    // lane-half g: K in {8g..8g+7, 16+8g..16+8g+7}
    Frag b;
    b.q[0] = *(const u32x4*)(base + 8 * g);
    b.q[1] = *(const u32x4*)(base + 16 + 8 * g);
    return b;
}

// ---------------------------------------------------------------------------
// 3) Pass A: rowmax_i(S) -> dinv_i = 1/(1 - rowmax + eps)
//    one workgroup per 64-row strip; A fragments register-resident
// ---------------------------------------------------------------------------
__global__ void __launch_bounds__(256) rowmax_kernel(
        const unsigned short* __restrict__ ppack,
        const unsigned short* __restrict__ tpack,
        float* __restrict__ dinv) {
    int img = blockIdx.x >> 6;
    int i0  = (blockIdx.x & 63) * 64;
    int tid = threadIdx.x;
    int wave = tid >> 5, lane = tid & 31;
    int h = lane & 15, g = lane >> 4;
    int rg = wave & 3, cg = wave >> 2;
    int m0 = i0 + 16 * rg;
    const unsigned short* Pi = ppack + (size_t)img * C_DIM * HW;
    const unsigned short* Ti = tpack + (size_t)img * C_DIM * HW;

    Frag a[KC];
#pragma unroll
    for (int kc = 0; kc < KC; ++kc)
        a[kc] = loadA(Pi + ((size_t)kc * HW + (m0 + h)) * 32, g);

    f32x8 rmax;
#pragma unroll
    for (int r = 0; r < 8; ++r) rmax[r] = -3.0e38f;

    for (int j0 = 0; j0 < HW; j0 += 32) {
        int n0 = j0 + 16 * cg;
        f32x8 acc = {};
#pragma unroll
        for (int kc = 0; kc < KC; ++kc) {
            Frag b = loadB(Ti + ((size_t)kc * HW + (n0 + h)) * 32, g);
            acc = __builtin_amdgcn_wmma_f32_16x16x32_bf16(
                      false, a[kc].v, false, b.v, (short)0, acc, false, false);
        }
#pragma unroll
        for (int r = 0; r < 8; ++r) rmax[r] = fmaxf(rmax[r], acc[r]);
    }

    __shared__ float red[2][64];
#pragma unroll
    for (int r = 0; r < 8; ++r) {
        float v = rmax[r];
        v = fmaxf(v, __shfl_xor(v, 1, 32));
        v = fmaxf(v, __shfl_xor(v, 2, 32));
        v = fmaxf(v, __shfl_xor(v, 4, 32));
        v = fmaxf(v, __shfl_xor(v, 8, 32));
        if (h == 0) red[cg][16 * rg + 8 * g + r] = v;
    }
    __syncthreads();
    if (tid < 64) {
        float smax = fmaxf(red[0][tid], red[1][tid]);
        float dmin = 1.0f - smax;
        dinv[img * HW + i0 + tid] = 1.0f / (dmin + CX_EPS);
    }
}

// ---------------------------------------------------------------------------
// 4) Pass B: rowsum_i( exp(1 - (1-S)*dinv_i) ) -> rsuminv_i
// ---------------------------------------------------------------------------
__global__ void __launch_bounds__(256) rowsum_kernel(
        const unsigned short* __restrict__ ppack,
        const unsigned short* __restrict__ tpack,
        const float* __restrict__ dinv,
        float* __restrict__ rsuminv) {
    int img = blockIdx.x >> 6;
    int i0  = (blockIdx.x & 63) * 64;
    int tid = threadIdx.x;
    int wave = tid >> 5, lane = tid & 31;
    int h = lane & 15, g = lane >> 4;
    int rg = wave & 3, cg = wave >> 2;
    int m0 = i0 + 16 * rg;
    const unsigned short* Pi = ppack + (size_t)img * C_DIM * HW;
    const unsigned short* Ti = tpack + (size_t)img * C_DIM * HW;

    Frag a[KC];
#pragma unroll
    for (int kc = 0; kc < KC; ++kc)
        a[kc] = loadA(Pi + ((size_t)kc * HW + (m0 + h)) * 32, g);

    float invr[8];
#pragma unroll
    for (int r = 0; r < 8; ++r) invr[r] = dinv[img * HW + m0 + 8 * g + r];

    f32x8 rsum = {};
    for (int j0 = 0; j0 < HW; j0 += 32) {
        int n0 = j0 + 16 * cg;
        f32x8 acc = {};
#pragma unroll
        for (int kc = 0; kc < KC; ++kc) {
            Frag b = loadB(Ti + ((size_t)kc * HW + (n0 + h)) * 32, g);
            acc = __builtin_amdgcn_wmma_f32_16x16x32_bf16(
                      false, a[kc].v, false, b.v, (short)0, acc, false, false);
        }
#pragma unroll
        for (int r = 0; r < 8; ++r)
            rsum[r] += __expf(1.0f - (1.0f - acc[r]) * invr[r]);
    }

    __shared__ float red[2][64];
#pragma unroll
    for (int r = 0; r < 8; ++r) {
        float v = rsum[r];
        v += __shfl_xor(v, 1, 32);
        v += __shfl_xor(v, 2, 32);
        v += __shfl_xor(v, 4, 32);
        v += __shfl_xor(v, 8, 32);
        if (h == 0) red[cg][16 * rg + 8 * g + r] = v;
    }
    __syncthreads();
    if (tid < 64) {
        float total = red[0][tid] + red[1][tid];
        rsuminv[img * HW + i0 + tid] = 1.0f / total;
    }
}

// ---------------------------------------------------------------------------
// 5) Pass C: colmax_j( w_ij * rsuminv_i );  B fragments register-resident
// ---------------------------------------------------------------------------
__global__ void __launch_bounds__(256) colmax_kernel(
        const unsigned short* __restrict__ ppack,
        const unsigned short* __restrict__ tpack,
        const float* __restrict__ dinv,
        const float* __restrict__ rsuminv,
        float* __restrict__ colmax) {
    int img = blockIdx.x >> 6;
    int j0  = (blockIdx.x & 63) * 64;
    int tid = threadIdx.x;
    int wave = tid >> 5, lane = tid & 31;
    int h = lane & 15, g = lane >> 4;
    int cg = wave & 3, rg = wave >> 2;
    int n0 = j0 + 16 * cg;
    const unsigned short* Pi = ppack + (size_t)img * C_DIM * HW;
    const unsigned short* Ti = tpack + (size_t)img * C_DIM * HW;

    Frag b[KC];
#pragma unroll
    for (int kc = 0; kc < KC; ++kc)
        b[kc] = loadB(Ti + ((size_t)kc * HW + (n0 + h)) * 32, g);

    float cmax = 0.0f;
    for (int i0 = 0; i0 < HW; i0 += 32) {
        int m0 = i0 + 16 * rg;
        f32x8 acc = {};
#pragma unroll
        for (int kc = 0; kc < KC; ++kc) {
            Frag a = loadA(Pi + ((size_t)kc * HW + (m0 + h)) * 32, g);
            acc = __builtin_amdgcn_wmma_f32_16x16x32_bf16(
                      false, a.v, false, b[kc].v, (short)0, acc, false, false);
        }
#pragma unroll
        for (int r = 0; r < 8; ++r) {
            int row = img * HW + m0 + 8 * g + r;
            float w = __expf(1.0f - (1.0f - acc[r]) * dinv[row]);
            cmax = fmaxf(cmax, w * rsuminv[row]);
        }
    }
    cmax = fmaxf(cmax, __shfl_xor(cmax, 16, 32));   // merge M-halves (same column)

    __shared__ float red[2][64];
    if (lane < 16) red[rg][16 * cg + h] = cmax;
    __syncthreads();
    if (tid < 64)
        colmax[img * HW + j0 + tid] = fmaxf(red[0][tid], red[1][tid]);
}

// ---------------------------------------------------------------------------
// 6) loss = mean_img( -log( mean_j colmax_j + eps ) )
// ---------------------------------------------------------------------------
__global__ void __launch_bounds__(256) final_kernel(const float* __restrict__ colmax,
                                                    float* __restrict__ out) {
    int tid = threadIdx.x;
    __shared__ float red[256];
    float total = 0.f;
    for (int img = 0; img < N_IMG; ++img) {
        float s = 0.f;
        for (int t = tid; t < HW; t += 256) s += colmax[img * HW + t];
        red[tid] = s; __syncthreads();
        for (int off = 128; off > 0; off >>= 1) {
            if (tid < off) red[tid] += red[tid + off];
            __syncthreads();
        }
        if (tid == 0) total += -logf(red[0] * (1.0f / HW) + CX_EPS);
        __syncthreads();
    }
    if (tid == 0) out[0] = total * (1.0f / N_IMG);
}

// ---------------------------------------------------------------------------
extern "C" void kernel_launch(void* const* d_in, const int* in_sizes, int n_in,
                              void* d_out, int out_size, void* d_ws, size_t ws_size,
                              hipStream_t stream) {
    (void)in_sizes; (void)n_in; (void)out_size; (void)ws_size;
    const float* pred   = (const float*)d_in[0];
    const float* target = (const float*)d_in[1];

    char* ws = (char*)d_ws;
    size_t packBytes = (size_t)N_IMG * C_DIM * HW * sizeof(unsigned short); // 16 MB
    unsigned short* ppack = (unsigned short*)ws;
    unsigned short* tpack = (unsigned short*)(ws + packBytes);
    float* mu      = (float*)(ws + 2 * packBytes);
    float* dinv    = mu + N_IMG * C_DIM;
    float* rsuminv = dinv + N_IMG * HW;
    float* colmax  = rsuminv + N_IMG * HW;
    float* out = (float*)d_out;

    mu_kernel      <<<N_IMG * C_DIM,      256, 0, stream>>>(target, mu);
    normpack_kernel<<<(N_IMG * HW) / 256, 256, 0, stream>>>(pred, target, mu, ppack, tpack);
    rowmax_kernel  <<<N_IMG * 64,         256, 0, stream>>>(ppack, tpack, dinv);
    rowsum_kernel  <<<N_IMG * 64,         256, 0, stream>>>(ppack, tpack, dinv, rsuminv);
    colmax_kernel  <<<N_IMG * 64,         256, 0, stream>>>(ppack, tpack, dinv, rsuminv, colmax);
    final_kernel   <<<1,                  256, 0, stream>>>(colmax, out);
}
